// VSA_layer_36867999268881
// MI455X (gfx1250) — matrix operand order repaired and checked by reading, not compiled, not run on hardware
//
#include <hip/hip_runtime.h>

typedef __attribute__((ext_vector_type(16))) _Float16 v16h;
typedef __attribute__((ext_vector_type(8)))  float    v8f;

#define BT_   96
#define NN_   400
#define DD_   256
#define HH_   8
#define HD_   32
#define MROWS_ (BT_*NN_)   // 38400
#define NPAD_ 416
#define QSCALE 0.17677669529663687f  // 1/sqrt(32)

union Frag16 { v16h h; uint4 q[2]; };
union Pack8  { _Float16 h[8]; uint4 q; };

// Scheduler hints (IGroupLP): mask 0x8=MFMA/WMMA, 0x20=VMEM_READ,
// 0x100=DS_READ, 0x200=DS_WRITE. SBAR() delimits a scheduling region so the
// group matcher cannot satisfy the hints by reordering across K-steps.
#if defined(__has_builtin)
#if __has_builtin(__builtin_amdgcn_sched_group_barrier)
#define SGB(mask, size) __builtin_amdgcn_sched_group_barrier(mask, size, 0)
#endif
#if __has_builtin(__builtin_amdgcn_sched_barrier)
#define SBAR() __builtin_amdgcn_sched_barrier(0)
#endif
#endif
#ifndef SGB
#define SGB(mask, size)
#endif
#ifndef SBAR
#define SBAR()
#endif

// Load a 16-element f16 fragment as two contiguous 16B chunks.
__device__ __forceinline__ v16h ld_frag(const _Float16* p0, const _Float16* p1) {
  Frag16 f;
  f.q[0] = *reinterpret_cast<const uint4*>(p0);
  f.q[1] = *reinterpret_cast<const uint4*>(p1);
  return f.h;
}

// ---------------- TDM support (toolchain-adaptive) ----------------
#if defined(__has_builtin)
#if __has_builtin(__builtin_amdgcn_tensor_load_to_lds) && \
    __has_builtin(__builtin_amdgcn_s_wait_tensorcnt)
#define HAVE_TDM 1
#if __has_include(<hip/amd_detail/amd_gfx1250_TDM.h>)
#define TDM_6ARG 1
#else
#define TDM_6ARG 0
#endif
#endif
#endif
#ifndef HAVE_TDM
#define HAVE_TDM 0
#endif

#if HAVE_TDM
typedef unsigned int u32x4_t __attribute__((ext_vector_type(4)));
typedef int          i32x4_t __attribute__((ext_vector_type(4)));
typedef int          i32x8_t __attribute__((ext_vector_type(8)));

// 2D TDM load, 2-byte elements: tile dim0 x dim1, row stride stride0 (elements).
// D# group0: count=1 | lds_addr | global_addr[56:0] | type=2 (bits 127:126).
// D# group1: data_size=1(2B) @17:16; tensor_dim0 @79:48; tensor_dim1 @111:80;
//            tile_dim0 @127:112; tile_dim1 @143:128; dim0_stride @207:160.
__device__ __forceinline__ void tdm_load_2d(unsigned lds_off,
                                            unsigned long long ga,
                                            unsigned dim0, unsigned dim1,
                                            unsigned stride0) {
  u32x4_t g0 = { 1u, lds_off,
                 (unsigned)(ga & 0xffffffffull),
                 (unsigned)(((ga >> 32) & 0x01ffffffull) | (2u << 30)) };
  i32x8_t g1;
  g1[0] = (int)(1u << 16);                                        // data_size=2B
  g1[1] = (int)((dim0 & 0xffffu) << 16);                          // tensor_dim0 lo
  g1[2] = (int)(((dim0 >> 16) & 0xffffu) | ((dim1 & 0xffffu) << 16)); // dim0 hi | dim1 lo
  g1[3] = (int)(((dim1 >> 16) & 0xffffu) | ((dim0 & 0xffffu) << 16)); // dim1 hi | tile_dim0
  g1[4] = (int)(dim1 & 0xffffu);                                  // tile_dim1 (tile_dim2=0)
  g1[5] = (int)stride0;                                           // dim0_stride lo32
  g1[6] = 0;
  g1[7] = 0;
  i32x4_t z = {0, 0, 0, 0};
#if TDM_6ARG
  i32x8_t z8 = {0, 0, 0, 0, 0, 0, 0, 0};
  __builtin_amdgcn_tensor_load_to_lds(g0, g1, z, z, z8, 0);
#else
  __builtin_amdgcn_tensor_load_to_lds(g0, g1, z, z, 0);
#endif
}
#endif

// ---------------- Kernel 0a: x (f32) -> xh (f16) ----------------
__global__ __launch_bounds__(256) void convert_x(const float* __restrict__ x,
                                                 _Float16* __restrict__ xh) {
  size_t i = ((size_t)blockIdx.x * 256 + threadIdx.x) * 4;
  float4 v = *reinterpret_cast<const float4*>(x + i);
  union { _Float16 h[4]; uint2 u; } pk;
  pk.h[0] = (_Float16)v.x; pk.h[1] = (_Float16)v.y;
  pk.h[2] = (_Float16)v.z; pk.h[3] = (_Float16)v.w;
  *reinterpret_cast<uint2*>(xh + i) = pk.u;
}

// ---------------- Kernel 0b: [Wq;Wk;Wv] (f32) -> wh (f16, [768][256]) ----------------
__global__ __launch_bounds__(256) void convert_w(const float* __restrict__ Wq,
                                                 const float* __restrict__ Wk,
                                                 const float* __restrict__ Wv,
                                                 _Float16* __restrict__ wh) {
  int i = (blockIdx.x * 256 + threadIdx.x) * 4;
  int j = i >> 8, k = i & 255;
  const float* src = (j < 256) ? (Wq + (size_t)j * 256)
                   : (j < 512) ? (Wk + (size_t)(j - 256) * 256)
                               : (Wv + (size_t)(j - 512) * 256);
  float4 v = *reinterpret_cast<const float4*>(src + k);
  union { _Float16 h[4]; uint2 u; } pk;
  pk.h[0] = (_Float16)v.x; pk.h[1] = (_Float16)v.y;
  pk.h[2] = (_Float16)v.z; pk.h[3] = (_Float16)v.w;
  *reinterpret_cast<uint2*>(wh + i) = pk.u;
}

// ---------------- Kernel 0c: zero vt padding columns m=400..415 ----------------
__global__ __launch_bounds__(256) void zero_vt_tail(_Float16* __restrict__ vt) {
  int r = blockIdx.x * 256 + threadIdx.x;   // 24576 rows = BT*H*hd
  uint4 z; z.x = z.y = z.z = z.w = 0u;
  uint4* p = reinterpret_cast<uint4*>(vt + (size_t)r * NPAD_ + NN_);
  p[0] = z; p[1] = z;
}

// ---------------- Kernel 1: fused QKV projection GEMM ----------------
// Block tile 256(M) x 64(N), 8 waves; each wave 32 rows x 64 cols = 2x4 WMMA
// accumulators. Per K-step region: 12 next-step VMEM reads issue first, then
// this step's 8 WMMAs (operands loaded one region earlier -> no wait-0 stalls).
__global__ __launch_bounds__(256) void qkv_gemm(
    const _Float16* __restrict__ xh, const _Float16* __restrict__ wh,
    const float* __restrict__ bq, const float* __restrict__ bk,
    const float* __restrict__ bv,
    _Float16* __restrict__ qh, _Float16* __restrict__ kh,
    _Float16* __restrict__ vt) {
  const int lane = threadIdx.x & 31;
  const int w    = threadIdx.x >> 5;
  const int hi   = lane >> 4;      // half-wave select
  const int ln   = lane & 15;
  const int m0   = blockIdx.x * 256 + w * 32;
  const int j0   = blockIdx.y * 64;            // global output column base
  const int matq = blockIdx.y >> 2;            // 0=Q 1=K 2=V (uniform per block)
  const int jl0  = j0 - matq * 256;            // column base within matrix

  v8f acc[2][4];
  #pragma unroll
  for (int u = 0; u < 2; ++u)
    #pragma unroll
    for (int t = 0; t < 4; ++t)
      #pragma unroll
      for (int r = 0; r < 8; ++r) acc[u][t][r] = 0.f;

  const _Float16* xrow0 = xh + (size_t)(m0 + ln) * DD_;
  const _Float16* xrow1 = xh + (size_t)(m0 + 16 + ln) * DD_;
  const _Float16* wrow0 = wh + (size_t)(j0 +  0 + ln) * DD_;
  const _Float16* wrow1 = wh + (size_t)(j0 + 16 + ln) * DD_;
  const _Float16* wrow2 = wh + (size_t)(j0 + 32 + ln) * DD_;
  const _Float16* wrow3 = wh + (size_t)(j0 + 48 + ln) * DD_;
  const int kao = hi ? 16 : 0;
  const int kbo = hi ? 16 : 0;

  v16h a0 = ld_frag(xrow0 + kao, xrow0 + kao + 8);
  v16h a1 = ld_frag(xrow1 + kao, xrow1 + kao + 8);
  v16h b0 = ld_frag(wrow0 + kbo, wrow0 + kbo + 8);
  v16h b1 = ld_frag(wrow1 + kbo, wrow1 + kbo + 8);
  v16h b2 = ld_frag(wrow2 + kbo, wrow2 + kbo + 8);
  v16h b3 = ld_frag(wrow3 + kbo, wrow3 + kbo + 8);
  SBAR();
  for (int kk = 0; kk < DD_; kk += 32) {
    const int kn  = (kk + 32) & (DD_ - 1);   // wraps to 0 on last iter (dead load)
    const int kan = kn + kao, kbn = kn + kbo;
    v16h an0 = ld_frag(xrow0 + kan, xrow0 + kan + 8);
    v16h an1 = ld_frag(xrow1 + kan, xrow1 + kan + 8);
    v16h bn0 = ld_frag(wrow0 + kbn, wrow0 + kbn + 8);
    v16h bn1 = ld_frag(wrow1 + kbn, wrow1 + kbn + 8);
    v16h bn2 = ld_frag(wrow2 + kbn, wrow2 + kbn + 8);
    v16h bn3 = ld_frag(wrow3 + kbn, wrow3 + kbn + 8);
    acc[0][0] = __builtin_amdgcn_wmma_f32_16x16x32_f16(false, a0, false, b0, (short)0, acc[0][0], false, false);
    acc[1][0] = __builtin_amdgcn_wmma_f32_16x16x32_f16(false, a1, false, b0, (short)0, acc[1][0], false, false);
    acc[0][1] = __builtin_amdgcn_wmma_f32_16x16x32_f16(false, a0, false, b1, (short)0, acc[0][1], false, false);
    acc[1][1] = __builtin_amdgcn_wmma_f32_16x16x32_f16(false, a1, false, b1, (short)0, acc[1][1], false, false);
    acc[0][2] = __builtin_amdgcn_wmma_f32_16x16x32_f16(false, a0, false, b2, (short)0, acc[0][2], false, false);
    acc[1][2] = __builtin_amdgcn_wmma_f32_16x16x32_f16(false, a1, false, b2, (short)0, acc[1][2], false, false);
    acc[0][3] = __builtin_amdgcn_wmma_f32_16x16x32_f16(false, a0, false, b3, (short)0, acc[0][3], false, false);
    acc[1][3] = __builtin_amdgcn_wmma_f32_16x16x32_f16(false, a1, false, b3, (short)0, acc[1][3], false, false);
    // Within this region: all 12 next-step loads first, then this step's WMMAs.
    SGB(0x20, 12);
    SGB(0x8, 8);
    SBAR();
    a0 = an0; a1 = an1; b0 = bn0; b1 = bn1; b2 = bn2; b3 = bn3;
  }

  const float* bias = (matq == 0) ? bq : (matq == 1) ? bk : bv;
  if (matq < 2) {
    _Float16* dst = (matq == 0) ? qh : kh;
    const float scale = (matq == 0) ? QSCALE : 1.0f;  // fold 1/sqrt(hd) into Q
    #pragma unroll
    for (int u = 0; u < 2; ++u)
      #pragma unroll
      for (int t = 0; t < 4; ++t) {
        const int jl = jl0 + t * 16 + ln;
        const float bb = bias[jl];
        #pragma unroll
        for (int r = 0; r < 8; ++r) {
          const int m = m0 + u * 16 + r + hi * 8;
          dst[(size_t)m * DD_ + jl] = (_Float16)((acc[u][t][r] + bb) * scale);
        }
      }
  } else {
    // V: store transposed vt[bt][h][dh][n], n padded to 416. 16-row tiles never
    // cross a bt boundary (400 % 16 == 0), so one packed b128 store per tile.
    #pragma unroll
    for (int u = 0; u < 2; ++u) {
      const int mfirst = m0 + u * 16 + hi * 8;
      const int bt = mfirst / NN_;
      const int n1 = mfirst - bt * NN_;
      #pragma unroll
      for (int t = 0; t < 4; ++t) {
        const int jl = jl0 + t * 16 + ln;      // 0..255
        const float bb = bias[jl];
        const int hh = jl >> 5, dh = jl & 31;
        Pack8 pk;
        #pragma unroll
        for (int r = 0; r < 8; ++r) pk.h[r] = (_Float16)(acc[u][t][r] + bb);
        *reinterpret_cast<uint4*>(
            vt + (size_t)((bt * HH_ + hh) * HD_ + dh) * NPAD_ + n1) = pk.q;
      }
    }
  }
}

// ---------------- Kernel 2: fused attention (scores * adj -> softmax -> P.V) ----------------
// One block per (n-tile, head, bt). 4 waves. Scores live entirely in LDS.
// The V-head slab (32x416 f16, contiguous) is DMA'd to LDS by the TDM at kernel
// entry and lands while QK^T + softmax run; phase 3 consumes it from LDS.
__global__ __launch_bounds__(128) void attn(
    const _Float16* __restrict__ qh, const _Float16* __restrict__ kh,
    const _Float16* __restrict__ vt, const float* __restrict__ adj,
    float* __restrict__ hbuf) {
  __shared__ __align__(16) float    s_s[16][NN_];     // 25.6 KB scores/exp
  __shared__ __align__(16) _Float16 s_p[16][NPAD_];   // 13.3 KB probs (f16, zero-padded)
  __shared__ __align__(16) float    s_acc[4][256];    // split-K partials
#if HAVE_TDM
  __shared__ __align__(16) _Float16 s_v[HD_][NPAD_];  // 26.6 KB V-head slab (TDM dest)
#endif

  const int tid  = threadIdx.x;
  const int lane = tid & 31;
  const int w    = tid >> 5;
  const int hi   = lane >> 4, ln = lane & 15;
  const int n0   = blockIdx.x * 16;
  const int h    = blockIdx.y;
  const int bt   = blockIdx.z;
  const size_t rowbase = (size_t)bt * NN_;

#if HAVE_TDM
  if (w == 0) {
    // Generic->LDS pointer truncation yields the LDS byte offset (LDS aperture
    // maps addr[31:0] to the wave's LDS space).
    tdm_load_2d((unsigned)(uintptr_t)&s_v[0][0],
                (unsigned long long)(uintptr_t)(vt + (size_t)(bt * HH_ + h) * HD_ * NPAD_),
                NPAD_, HD_, NPAD_);
  }
#endif

  // Q fragment for this wave (rows n0..n0+15; already scaled by 1/sqrt(hd))
  const _Float16* qrow = qh + (rowbase + n0 + ln) * DD_ + h * HD_;
  const int ka = hi ? 16 : 0;
  v16h aq = ld_frag(qrow + ka, qrow + ka + 8);

  // Phase 1: scores = (q.k) * adj, one WMMA per 16x16 tile (K == hd == 32).
  // K fragments are preloaded one m-tile ahead.
  {
    const _Float16* kbase = kh + rowbase * DD_ + h * HD_ + (hi ? 16 : 0) + (size_t)ln * DD_;
    v16h bk = ld_frag(kbase + (size_t)(w * 16) * DD_, kbase + (size_t)(w * 16) * DD_ + 8);
    SBAR();
    for (int mt = w; mt < 25; mt += 4) {
      const int mtn = (mt + 4 < 25) ? mt + 4 : mt;
      const _Float16* kn = kbase + (size_t)(mtn * 16) * DD_;
      v16h bkn = ld_frag(kn, kn + 8);
      v8f c;
      #pragma unroll
      for (int r = 0; r < 8; ++r) c[r] = 0.f;
      c = __builtin_amdgcn_wmma_f32_16x16x32_f16(
          false, aq, false, bk, (short)0, c, false, false);
      const int mcol = mt * 16 + ln;
      #pragma unroll
      for (int r = 0; r < 8; ++r) {
        const int nr = r + hi * 8;
        const float av = adj[(rowbase + n0 + nr) * NN_ + mcol];
        s_s[nr][mcol] = c[r] * av;
      }
      SGB(0x20, 10);  // next K fragment + this tile's adj loads
      SGB(0x8, 1);    // this tile's WMMA
      SGB(0x200, 8);  // score stores to LDS
      SBAR();
      bk = bkn;
    }
  }
  __syncthreads();

  // Phase 2: softmax over m (8 lanes per row, shuffle reductions in-wave)
  {
    const int r = tid >> 3, sub = tid & 7;
    float mx = -3.4e38f;
    for (int m = sub; m < NN_; m += 8) mx = fmaxf(mx, s_s[r][m]);
    for (int msk = 1; msk < 8; msk <<= 1) mx = fmaxf(mx, __shfl_xor(mx, msk, 8));
    float sm = 0.f;
    for (int m = sub; m < NN_; m += 8) {
      float e = __expf(s_s[r][m] - mx);
      s_s[r][m] = e;
      sm += e;
    }
    for (int msk = 1; msk < 8; msk <<= 1) sm += __shfl_xor(sm, msk, 8);
    const float rinv = 1.f / sm;
    for (int m = sub; m < NN_; m += 8) s_p[r][m] = (_Float16)(s_s[r][m] * rinv);
    for (int m = NN_ + sub; m < NPAD_; m += 8) s_p[r][m] = (_Float16)0.f;
  }
#if HAVE_TDM
  if (w == 0) __builtin_amdgcn_s_wait_tensorcnt(0);  // V slab resident in LDS
#endif
  __syncthreads();

  // Phase 3: out = P x V (split-K across wave pairs; vt padding cols are zeroed
  // in P so any vt tail garbage contributes nothing). Pipelined fragment loads.
  {
    const int tdh = w & 1;
    v8f c;
    #pragma unroll
    for (int r = 0; r < 8; ++r) c[r] = 0.f;
#if HAVE_TDM
    const _Float16* vrow = &s_v[tdh * 16 + ln][0];
#else
    const _Float16* vrow =
        vt + (size_t)((bt * HH_ + h) * HD_ + tdh * 16 + ln) * NPAD_;
#endif
    const int c0 = w >> 1;
    const _Float16* pr0 = &s_p[ln][c0 * 32 + (hi ? 16 : 0)];
    const _Float16* vr0 = vrow + c0 * 32 + (hi ? 16 : 0);
    v16h ap = ld_frag(pr0, pr0 + 8);
    v16h bv = ld_frag(vr0, vr0 + 8);
    SBAR();
    for (int cch = c0; cch < 13; cch += 2) {
      const int cn = (cch + 2 < 13) ? cch + 2 : cch;
      const _Float16* prn = &s_p[ln][cn * 32 + (hi ? 16 : 0)];
      const _Float16* vrn = vrow + cn * 32 + (hi ? 16 : 0);
      v16h apn = ld_frag(prn, prn + 8);
      v16h bvn = ld_frag(vrn, vrn + 8);
      c = __builtin_amdgcn_wmma_f32_16x16x32_f16(
          false, ap, false, bv, (short)0, c, false, false);
      SGB(0x120, 4);  // next P/V fragments (DS_READ | VMEM_READ)
      SGB(0x8, 1);    // this chunk's WMMA
      SBAR();
      ap = apn; bv = bvn;
    }
    #pragma unroll
    for (int r = 0; r < 8; ++r) s_acc[w][(r + hi * 8) * 16 + ln] = c[r];
  }
  __syncthreads();

  for (int o = tid; o < 512; o += 128) {
    const int n = o >> 5, dh = o & 31;
    const int tl = dh >> 4, e = (n << 4) | (dh & 15);
    const float v = s_acc[tl][e] + s_acc[tl + 2][e];
    hbuf[(rowbase + n0 + n) * DD_ + h * HD_ + dh] = v;
  }
}

// ---------------- Kernel 3: residual + LayerNorm (one wave per row) ----------------
__global__ __launch_bounds__(256) void resid_ln(
    const float* __restrict__ x, const float* __restrict__ hbuf,
    const float* __restrict__ gamma, const float* __restrict__ beta,
    float* __restrict__ out) {
  const int lane = threadIdx.x & 31;
  const int w    = threadIdx.x >> 5;
  const int m    = blockIdx.x * 8 + w;
  const size_t base = (size_t)m * DD_ + lane * 8;
  float4 a0 = *reinterpret_cast<const float4*>(x + base);
  float4 a1 = *reinterpret_cast<const float4*>(x + base + 4);
  float4 b0 = *reinterpret_cast<const float4*>(hbuf + base);
  float4 b1 = *reinterpret_cast<const float4*>(hbuf + base + 4);
  float y[8] = {a0.x + b0.x, a0.y + b0.y, a0.z + b0.z, a0.w + b0.w,
                a1.x + b1.x, a1.y + b1.y, a1.z + b1.z, a1.w + b1.w};
  float s = 0.f;
  #pragma unroll
  for (int i = 0; i < 8; ++i) s += y[i];
  for (int msk = 1; msk < 32; msk <<= 1) s += __shfl_xor(s, msk, 32);
  const float mu = s * (1.f / DD_);
  float v = 0.f;
  #pragma unroll
  for (int i = 0; i < 8; ++i) { float d = y[i] - mu; v += d * d; }
  for (int msk = 1; msk < 32; msk <<= 1) v += __shfl_xor(v, msk, 32);
  const float rs = rsqrtf(v * (1.f / DD_) + 1e-5f);
  float4 g0 = *reinterpret_cast<const float4*>(gamma + lane * 8);
  float4 g1 = *reinterpret_cast<const float4*>(gamma + lane * 8 + 4);
  float4 t0 = *reinterpret_cast<const float4*>(beta + lane * 8);
  float4 t1 = *reinterpret_cast<const float4*>(beta + lane * 8 + 4);
  float4 o0, o1;
  o0.x = (y[0] - mu) * rs * g0.x + t0.x;
  o0.y = (y[1] - mu) * rs * g0.y + t0.y;
  o0.z = (y[2] - mu) * rs * g0.z + t0.z;
  o0.w = (y[3] - mu) * rs * g0.w + t0.w;
  o1.x = (y[4] - mu) * rs * g1.x + t1.x;
  o1.y = (y[5] - mu) * rs * g1.y + t1.y;
  o1.z = (y[6] - mu) * rs * g1.z + t1.z;
  o1.w = (y[7] - mu) * rs * g1.w + t1.w;
  *reinterpret_cast<float4*>(out + base) = o0;
  *reinterpret_cast<float4*>(out + base + 4) = o1;
}

extern "C" void kernel_launch(void* const* d_in, const int* in_sizes, int n_in,
                              void* d_out, int out_size, void* d_ws, size_t ws_size,
                              hipStream_t stream) {
  (void)in_sizes; (void)n_in; (void)out_size; (void)ws_size;
  const float* x     = (const float*)d_in[0];
  const float* adj   = (const float*)d_in[1];
  const float* Wq    = (const float*)d_in[2];
  const float* bq    = (const float*)d_in[3];
  const float* Wk    = (const float*)d_in[4];
  const float* bk    = (const float*)d_in[5];
  const float* Wv    = (const float*)d_in[6];
  const float* bv    = (const float*)d_in[7];
  const float* gamma = (const float*)d_in[8];
  const float* beta  = (const float*)d_in[9];
  float* out = (float*)d_out;

  char* ws = (char*)d_ws;
  size_t off = 0;
  _Float16* xh = (_Float16*)(ws + off); off += (size_t)MROWS_ * DD_ * 2;
  _Float16* wh = (_Float16*)(ws + off); off += (size_t)768 * DD_ * 2;
  _Float16* qh = (_Float16*)(ws + off); off += (size_t)MROWS_ * DD_ * 2;
  _Float16* kh = (_Float16*)(ws + off); off += (size_t)MROWS_ * DD_ * 2;
  _Float16* vt = (_Float16*)(ws + off); off += (size_t)BT_ * HH_ * HD_ * NPAD_ * 2;
  float*  hbuf = (float*)(ws + off);    off += (size_t)MROWS_ * DD_ * 4;

  convert_x<<<9600, 256, 0, stream>>>(x, xh);                 // 38400*256 elems
  convert_w<<<192, 256, 0, stream>>>(Wq, Wk, Wv, wh);         // 768*256 elems
  zero_vt_tail<<<96, 256, 0, stream>>>(vt);                   // 24576 rows
  qkv_gemm<<<dim3(150, 12), 256, 0, stream>>>(xh, wh, bq, bk, bv, qh, kh, vt);
  attn<<<dim3(25, 8, 96), 128, 0, stream>>>(qh, kh, vt, adj, hbuf);
  resid_ln<<<4800, 256, 0, stream>>>(x, hbuf, gamma, beta, out);
}
